// VanillaRNN_68212670595843
// MI455X (gfx1250) — compile-verified, matrix operand order checked by reading, not run on hardware
//
#include <hip/hip_runtime.h>
#include <hip/hip_bf16.h>
#include <cstdint>

// ---------------------------------------------------------------------------
// CDNA5 (gfx1250) bf16 WMMA RNN: xh = x@Wxh^T + bxh (one big GEMM),
// h_t = tanh(xh_t + h@Whh^T + bhh) (256 serial step GEMMs),
// out = h_T@Why^T + bhy.
// Double-buffered LDS; async global->LDS copies (ASYNCcnt) when available.
// ---------------------------------------------------------------------------

typedef __attribute__((ext_vector_type(16))) __bf16 v16bf;
typedef __attribute__((ext_vector_type(8)))  float  v8f;
typedef __attribute__((ext_vector_type(4)))  int    v4i;

#define LDP 40   // LDS row pitch in halves (32 data + 8 pad -> 80B pitch)

#if defined(__AMDGCN__) && \
    __has_builtin(__builtin_amdgcn_global_load_async_to_lds_b128) && \
    __has_builtin(__builtin_amdgcn_s_wait_asynccnt)
#define USE_ASYNC_LDS 1
#else
#define USE_ASYNC_LDS 0
#endif

// Pointer types matching the async-load builtin's parameters:
//   arg0: int4 in addrspace(1) (global), arg1: int4 in addrspace(3) (LDS)
typedef __attribute__((address_space(1))) v4i* gv4i_p;
typedef __attribute__((address_space(3))) v4i* lv4i_p;

__device__ __forceinline__ unsigned short f32_to_bf16_rne(float f) {
  unsigned int u = __float_as_uint(f);
  u += 0x7FFFu + ((u >> 16) & 1u);           // round to nearest even
  return (unsigned short)(u >> 16);
}
__device__ __forceinline__ float bf16_to_f32(unsigned short h) {
  return __uint_as_float(((unsigned int)h) << 16);
}

// ---------------------------------------------------------------------------
// Tiled bf16 GEMM with f32 accumulation:
//   C[M,N] = A[M,K] * Bw[N,K]^T  (Bw holds weight rows; B-fragment == A-layout of Bw)
// Optional: + bias[N], + addB[M,N] (bf16), tanh, write f32 and/or bf16.
// Block = 256 threads = 8 waves. Block tile 128x128. Wave tile 32x64.
// Requires M%128==0, N%128==0, K%32==0 (true for all call sites here).
// ---------------------------------------------------------------------------
__global__ __launch_bounds__(256)
void rnn_wmma_gemm(const unsigned short* __restrict__ A,
                   const unsigned short* __restrict__ Bw,
                   const float* __restrict__ bias,
                   const unsigned short* __restrict__ addB,
                   float* __restrict__ outF,
                   unsigned short* __restrict__ outB,
                   int M, int N, int K, int act)
{
  __shared__ unsigned short Alds[2][128][LDP];
  __shared__ unsigned short Blds[2][128][LDP];

  const int tid   = threadIdx.x;
  const int lane  = tid & 31;
  const int wave  = tid >> 5;
  const int waveM = wave >> 1;     // 0..3 -> 32-row slab
  const int waveN = wave & 1;      // 0..1 -> 64-col slab
  const int rowBase = blockIdx.y * 128;
  const int colBase = blockIdx.x * 128;

  const int lr = lane & 15;        // row within 16x16 tile
  const int kg = (lane >> 4) & 1;  // K-group: lanes16-31 take K 8..15 / 24..31

  v8f acc[2][4];
#pragma unroll
  for (int mi = 0; mi < 2; ++mi)
#pragma unroll
    for (int ni = 0; ni < 4; ++ni)
      acc[mi][ni] = {};

  // Stage one 128x32 K-slab of A and B into LDS buffer `buf`.
  // Per thread: 2 chunks of A + 2 chunks of B (16B each).
  // Per wave: 4 async-load instructions -> ASYNCcnt += 4.
  auto stage = [&](int it, int buf) {
#pragma unroll
    for (int i = 0; i < 2; ++i) {
      int q = tid + (i << 8);          // 0..511 -> 128 rows x 4 chunks of 8 halves
      int r = q >> 2;
      int c = (q & 3) << 3;
      size_t ka = (size_t)(rowBase + r) * K + (it * 32 + c);
      size_t kb = (size_t)(colBase + r) * K + (it * 32 + c);
#if USE_ASYNC_LDS
      __builtin_amdgcn_global_load_async_to_lds_b128(
          (gv4i_p)(A + ka), (lv4i_p)&Alds[buf][r][c], 0, 0);
      __builtin_amdgcn_global_load_async_to_lds_b128(
          (gv4i_p)(Bw + kb), (lv4i_p)&Blds[buf][r][c], 0, 0);
#else
      *(uint4*)(&Alds[buf][r][c]) = *(const uint4*)(A + ka);
      *(uint4*)(&Blds[buf][r][c]) = *(const uint4*)(Bw + kb);
#endif
    }
  };

  // LDS fragment loader matching the ISA 16-bit A/B layout:
  // chunk0 = K[kg*8 .. kg*8+7], chunk1 = K[16+kg*8 .. 16+kg*8+7]
  auto ldfrag = [&](unsigned short (*tile)[LDP], int row) -> v16bf {
    const unsigned short* p0 = &tile[row + lr][kg * 8];
    union { uint4 u; __bf16 b[8]; } c0, c1;
    c0.u = *(const uint4*)p0;          // ds_load_b128
    c1.u = *(const uint4*)(p0 + 16);   // ds_load_b128
    v16bf f;
#pragma unroll
    for (int i = 0; i < 8; ++i) { f[i] = c0.b[i]; f[i + 8] = c1.b[i]; }
    return f;
  };

  const int nIter = K >> 5;
  stage(0, 0);

  for (int it = 0; it < nIter; ++it) {
    const int buf = it & 1;
    if (it + 1 < nIter) {
      stage(it + 1, buf ^ 1);          // next slab in flight while we compute
#if USE_ASYNC_LDS
      __builtin_amdgcn_s_wait_asynccnt(4);   // oldest 4 (current buf) landed
#endif
    } else {
#if USE_ASYNC_LDS
      __builtin_amdgcn_s_wait_asynccnt(0);
#else
      ;
#endif
    }
#if !USE_ASYNC_LDS
    if (it + 1 < nIter) {              // warm L2 for the next slab (it+1)
      __builtin_prefetch((const void*)(A  + (size_t)(rowBase + (tid >> 1)) * K + (it * 32 + 32)), 0, 1);
      __builtin_prefetch((const void*)(Bw + (size_t)(colBase + (tid >> 1)) * K + (it * 32 + 32)), 0, 1);
    }
#endif
    __syncthreads();

    // --- fragments + 8 WMMAs per wave ------------------------------------
    v16bf afrag[2], bfrag[4];
    afrag[0] = ldfrag(Alds[buf], waveM * 32);
    afrag[1] = ldfrag(Alds[buf], waveM * 32 + 16);
#pragma unroll
    for (int ni = 0; ni < 4; ++ni)
      bfrag[ni] = ldfrag(Blds[buf], waveN * 64 + ni * 16);

#pragma unroll
    for (int mi = 0; mi < 2; ++mi)
#pragma unroll
      for (int ni = 0; ni < 4; ++ni)
        acc[mi][ni] = __builtin_amdgcn_wmma_f32_16x16x32_bf16(
            false, afrag[mi], false, bfrag[ni], (short)0, acc[mi][ni],
            false, false);

    __syncthreads();                   // everyone done reading buf before re-stage
  }

  // --- epilogue: bias + residual(addB) + tanh, write f32/bf16 -------------
  const int hi = lane >> 4;
#pragma unroll
  for (int mi = 0; mi < 2; ++mi) {
#pragma unroll
    for (int ni = 0; ni < 4; ++ni) {
      const int col = colBase + waveN * 64 + ni * 16 + lr;
      const float bv = bias ? bias[col] : 0.0f;
#pragma unroll
      for (int v = 0; v < 8; ++v) {
        const int row = rowBase + waveM * 32 + mi * 16 + hi * 8 + v;
        const size_t idx = (size_t)row * N + col;
        float val = acc[mi][ni][v] + bv;
        if (addB) val += bf16_to_f32(addB[idx]);
        if (act)  val = tanhf(val);
        if (outF) outF[idx] = val;
        if (outB) outB[idx] = f32_to_bf16_rne(val);
      }
    }
  }
}

// f32 -> bf16 cast, 8 elements per thread (float4 x2 in, uint4 out)
__global__ __launch_bounds__(256)
void cast_f32_to_bf16_x8(const float4* __restrict__ in, uint4* __restrict__ out, long n8)
{
  long i = (long)blockIdx.x * blockDim.x + threadIdx.x;
  if (i >= n8) return;
  float4 a = in[2 * i];
  float4 b = in[2 * i + 1];
  union { uint4 u; unsigned short s[8]; } o;
  o.s[0] = f32_to_bf16_rne(a.x); o.s[1] = f32_to_bf16_rne(a.y);
  o.s[2] = f32_to_bf16_rne(a.z); o.s[3] = f32_to_bf16_rne(a.w);
  o.s[4] = f32_to_bf16_rne(b.x); o.s[5] = f32_to_bf16_rne(b.y);
  o.s[6] = f32_to_bf16_rne(b.z); o.s[7] = f32_to_bf16_rne(b.w);
  out[i] = o.u;
}

extern "C" void kernel_launch(void* const* d_in, const int* in_sizes, int n_in,
                              void* d_out, int out_size, void* d_ws, size_t ws_size,
                              hipStream_t stream)
{
  (void)in_sizes; (void)n_in; (void)out_size; (void)ws_size;

  const float* x   = (const float*)d_in[0];   // [T,B,I]
  const float* Wxh = (const float*)d_in[1];   // [H,I]
  const float* bxh = (const float*)d_in[2];   // [H]
  const float* Whh = (const float*)d_in[3];   // [H,H]
  const float* bhh = (const float*)d_in[4];   // [H]
  const float* Why = (const float*)d_in[5];   // [O,H]
  const float* bhy = (const float*)d_in[6];   // [O]
  float* out = (float*)d_out;                 // [B,O] f32

  const int T = 256, Bb = 256, I = 1024, H = 2048, O = 1024;
  const long MB = (long)T * Bb;               // 65536 rows of the big GEMM

  // Workspace layout (bf16 stores as ushort):
  unsigned short* xb    = (unsigned short*)d_ws;            // [MB, I]   128 MB
  unsigned short* Wxh_b = xb    + (size_t)MB * I;           // [H, I]      4 MB
  unsigned short* Whh_b = Wxh_b + (size_t)H * I;            // [H, H]      8 MB
  unsigned short* Why_b = Whh_b + (size_t)H * H;            // [O, H]      4 MB
  unsigned short* xh_b  = Why_b + (size_t)O * H;            // [MB, H]   256 MB
  unsigned short* h0    = xh_b  + (size_t)MB * H;           // [Bb, H]     1 MB
  unsigned short* h1    = h0    + (size_t)Bb * H;           // [Bb, H]     1 MB

  auto cast = [&](const float* src, unsigned short* dst, long n) {
    long n8 = n / 8;
    unsigned int g = (unsigned int)((n8 + 255) / 256);
    cast_f32_to_bf16_x8<<<dim3(g), dim3(256), 0, stream>>>(
        (const float4*)src, (uint4*)dst, n8);
  };
  cast(x,   xb,    MB * (long)I);
  cast(Wxh, Wxh_b, (long)H * I);
  cast(Whh, Whh_b, (long)H * H);
  cast(Why, Why_b, (long)O * H);

  // h_0 = 0 (bf16 zeros == bit zeros)
  (void)hipMemsetAsync(h0, 0, (size_t)Bb * H * sizeof(unsigned short), stream);

  // Phase 1: xh = x @ Wxh^T + bxh  (full-occupancy GEMM, 8192 workgroups)
  rnn_wmma_gemm<<<dim3(H / 128, MB / 128), dim3(256), 0, stream>>>(
      xb, Wxh_b, bxh, nullptr, nullptr, xh_b, (int)MB, H, I, 0);

  // Phase 2: 256 serial steps, Whh (8MB bf16) stays L2-resident
  unsigned short* hc = h0;
  unsigned short* hn = h1;
  for (int t = 0; t < T; ++t) {
    rnn_wmma_gemm<<<dim3(H / 128, Bb / 128), dim3(256), 0, stream>>>(
        hc, Whh_b, bhh, xh_b + (size_t)t * Bb * H, nullptr, hn, Bb, H, H, 1);
    unsigned short* tmp = hc; hc = hn; hn = tmp;
  }

  // Phase 3: out = h_T @ Why^T + bhy  (f32 output)
  rnn_wmma_gemm<<<dim3(O / 128, Bb / 128), dim3(256), 0, stream>>>(
      hc, Why_b, bhy, nullptr, out, nullptr, Bb, O, H, 0);
}
